// ContrastiveLoss_43559558316495
// MI455X (gfx1250) — matrix operand order repaired and checked by reading, not compiled, not run on hardware
//
#include <hip/hip_runtime.h>
#include <hip/hip_bf16.h>
#include <math.h>
#include <stdint.h>

typedef __attribute__((ext_vector_type(16))) _Float16 v16h;
typedef __attribute__((ext_vector_type(8)))  float    v8f;
typedef __attribute__((ext_vector_type(4)))  float    f32x4;

#define DD 256
#define EPSF 1e-8f
#define COLS_PER_CHUNK 512
#define STAGE_COLS 32
#define STAGES (COLS_PER_CHUNK / STAGE_COLS)          // 16
#define STAGE_BYTES (STAGE_COLS * DD * 2)             // 16384
#define STAGE_F32X4 (STAGE_BYTES / 16)                // 1024

union V16H { v16h v; f32x4 f4[2]; };
union V8F  { v8f  v; float f[8]; };

// ---------------------------------------------------------------------------
// Kernel 1: row-normalize x and y to f16, accumulate -(pos_xx+pos_xy+pos_yy)
// ---------------------------------------------------------------------------
__global__ void __launch_bounds__(256) normalize_kernel(
    const float* __restrict__ x, const float* __restrict__ y,
    _Float16* __restrict__ xn, _Float16* __restrict__ yn,
    float* __restrict__ out)
{
    const int n = blockIdx.x;
    const int t = threadIdx.x;
    const float xv = x[(size_t)n * DD + t];
    const float yv = y[(size_t)n * DD + t];
    float sx = xv * xv, sy = yv * yv, sxy = xv * yv;
    #pragma unroll
    for (int m = 16; m >= 1; m >>= 1) {
        sx  += __shfl_xor(sx,  m, 32);
        sy  += __shfl_xor(sy,  m, 32);
        sxy += __shfl_xor(sxy, m, 32);
    }
    __shared__ float sh[3][8];
    __shared__ float tot[3];
    const int w = t >> 5;
    if ((t & 31) == 0) { sh[0][w] = sx; sh[1][w] = sy; sh[2][w] = sxy; }
    __syncthreads();
    if (t == 0) {
        float a = 0.f, b = 0.f, c = 0.f;
        #pragma unroll
        for (int i = 0; i < 8; ++i) { a += sh[0][i]; b += sh[1][i]; c += sh[2][i]; }
        tot[0] = a; tot[1] = b; tot[2] = c;
    }
    __syncthreads();
    const float nx = sqrtf(tot[0]) + EPSF;
    const float ny = sqrtf(tot[1]) + EPSF;
    xn[(size_t)n * DD + t] = (_Float16)(xv / nx);
    yn[(size_t)n * DD + t] = (_Float16)(yv / ny);
    if (t == 0) {
        const float pos = tot[0] / (nx * nx) + tot[2] / (nx * ny) + tot[1] / (ny * ny);
        atomicAdd(out, -pos);
    }
}

// ---------------------------------------------------------------------------
// Async DMA of this thread's 64-byte slice of a 32-column B stage into LDS.
// INST_OFFSET applies to both the LDS and global address (ISA 08 §4.4), so
// one LDS addr + one global addr + offset immediates covers all 4 B128s.
// ---------------------------------------------------------------------------
__device__ __forceinline__ void async_stage(const _Float16* __restrict__ B,
                                            uint32_t ldsaddr, int colBase, int t)
{
    const uint64_t g = (uint64_t)(size_t)(B + (size_t)(colBase + (t >> 3)) * DD
                                          + (t & 7) * 32);
    asm volatile(
        "global_load_async_to_lds_b128 %0, %1, off\n\t"
        "global_load_async_to_lds_b128 %0, %1, off offset:16\n\t"
        "global_load_async_to_lds_b128 %0, %1, off offset:32\n\t"
        "global_load_async_to_lds_b128 %0, %1, off offset:48"
        :: "v"(ldsaddr), "v"(g) : "memory");
}

// ---------------------------------------------------------------------------
// Kernel 2: streaming sum-of-exp of similarity rows via WMMA f16->f32.
// grid = (N/COLS_PER_CHUNK, N/128, 3). Block = 256 threads (8 waves);
// each wave owns a 16-row A tile; B stages are double-buffered in LDS via
// global_load_async_to_lds_b128 (ASYNCcnt) so DMA overlaps WMMA.
// ---------------------------------------------------------------------------
__global__ void __launch_bounds__(256) sim_rowsum_kernel(
    const _Float16* __restrict__ xn, const _Float16* __restrict__ yn,
    float* __restrict__ S, int N)
{
    const int sim = blockIdx.z;
    const _Float16* __restrict__ A = (sim == 2) ? yn : xn;
    const _Float16* __restrict__ B = (sim == 0) ? xn : yn;

    const int t    = threadIdx.x;
    const int lane = t & 31;
    const int wave = t >> 5;
    const int hf   = lane >> 4;
    const int l16  = lane & 15;
    const int rowBase = blockIdx.y * 128 + wave * 16;
    const int m  = rowBase + l16;
    const int c0 = blockIdx.x * COLS_PER_CHUNK;

    // Resident A tile: 16 rows x 256 K as 8 chained 16x32 f16 operands.
    V16H a[8];
    #pragma unroll
    for (int kc = 0; kc < 8; ++kc) {
        const _Float16* ap = A + (size_t)m * DD + kc * 32 + hf * 8;
        a[kc].f4[0] = *(const f32x4*)(ap);
        a[kc].f4[1] = *(const f32x4*)(ap + 16);
    }

    __shared__ f32x4 ldsB[2][STAGE_F32X4];   // 2 x 16 KB double buffer
    const uint32_t ldsBase = (uint32_t)(size_t)(&ldsB[0][0]);

    V8F acc;
    #pragma unroll
    for (int r = 0; r < 8; ++r) acc.f[r] = 0.f;

    // Prologue: DMA stage 0 into buffer 0.
    async_stage(B, ldsBase + t * 64, c0, t);

    for (int s = 0; s < STAGES; ++s) {
        // Issue DMA for the next stage (redundant re-issue on the last
        // iteration keeps the wait threshold a constant immediate).
        const int ns = (s + 1 < STAGES) ? s + 1 : s;
        async_stage(B, ldsBase + ((s + 1) & 1) * STAGE_BYTES + t * 64,
                    c0 + ns * STAGE_COLS, t);
        // <=4 outstanding == everything except the just-issued next stage.
        asm volatile("s_wait_asynccnt 0x4" ::: "memory");
        __syncthreads();

        const f32x4* lb = &ldsB[s & 1][0];
        #pragma unroll
        for (int tt = 0; tt < 2; ++tt) {
            const int cb = (tt * 16 + l16) * 32 + hf;
            // Fetch all 8 B fragments first, then a back-to-back WMMA chain.
            V16H b[8];
            #pragma unroll
            for (int kc = 0; kc < 8; ++kc) {
                b[kc].f4[0] = lb[cb + kc * 4];
                b[kc].f4[1] = lb[cb + kc * 4 + 2];
            }
            V8F c;
            #pragma unroll
            for (int r = 0; r < 8; ++r) c.f[r] = 0.f;
            #pragma unroll
            for (int kc = 0; kc < 8; ++kc) {
                c.v = __builtin_amdgcn_wmma_f32_16x16x32_f16(
                          false, a[kc].v, false, b[kc].v, (short)0, c.v,
                          false, false);
            }
            // |sim| <= 1 after normalization -> unstabilized exp is safe.
            #pragma unroll
            for (int r = 0; r < 8; ++r) acc.f[r] += __expf(c.f[r]);
        }
        __syncthreads();
    }

    // C/D layout: VGPR r holds row (r + 8*hf); columns striped over 16 lanes.
    #pragma unroll
    for (int msk = 8; msk >= 1; msk >>= 1) {
        #pragma unroll
        for (int r = 0; r < 8; ++r) acc.f[r] += __shfl_xor(acc.f[r], msk, 32);
    }
    if (l16 == 0) {
        const int row0 = rowBase + hf * 8;
        #pragma unroll
        for (int r = 0; r < 8; ++r)
            atomicAdd(&S[(size_t)sim * N + row0 + r], acc.f[r]);
    }
}

// ---------------------------------------------------------------------------
// Kernel 3: out += sum_i log(S[i])
// ---------------------------------------------------------------------------
__global__ void __launch_bounds__(256) finish_kernel(
    const float* __restrict__ S, float* __restrict__ out, int total)
{
    const int idx = blockIdx.x * 256 + threadIdx.x;
    float v = (idx < total) ? logf(S[idx]) : 0.f;
    #pragma unroll
    for (int m = 16; m >= 1; m >>= 1) v += __shfl_xor(v, m, 32);
    __shared__ float sh[8];
    if ((threadIdx.x & 31) == 0) sh[threadIdx.x >> 5] = v;
    __syncthreads();
    if (threadIdx.x == 0) {
        float s = 0.f;
        #pragma unroll
        for (int i = 0; i < 8; ++i) s += sh[i];
        atomicAdd(out, s);
    }
}

extern "C" void kernel_launch(void* const* d_in, const int* in_sizes, int n_in,
                              void* d_out, int out_size, void* d_ws, size_t ws_size,
                              hipStream_t stream)
{
    const float* x = (const float*)d_in[0];
    const float* y = (const float*)d_in[1];
    const int N = in_sizes[0] / DD;   // 8192
    float* out = (float*)d_out;

    char* ws = (char*)d_ws;
    _Float16* xn = (_Float16*)ws;
    _Float16* yn = (_Float16*)(ws + (size_t)N * DD * sizeof(_Float16));
    float*    S  = (float*)   (ws + (size_t)N * DD * sizeof(_Float16) * 2);

    hipMemsetAsync(out, 0, sizeof(float), stream);
    hipMemsetAsync(S, 0, (size_t)3 * N * sizeof(float), stream);

    normalize_kernel<<<N, 256, 0, stream>>>(x, y, xn, yn, out);

    dim3 grid(N / COLS_PER_CHUNK, N / 128, 3);
    sim_rowsum_kernel<<<grid, 256, 0, stream>>>(xn, yn, S, N);

    finish_kernel<<<(3 * N + 255) / 256, 256, 0, stream>>>(S, out, 3 * N);
}
